// GraphLayer_50087908606288
// MI455X (gfx1250) — matrix operand order repaired and checked by reading, not compiled, not run on hardware
//
#include <hip/hip_runtime.h>
#include <hip/hip_bf16.h>
#include <math.h>

// ---------------------------------------------------------------------------
// Problem constants (match reference)
// ---------------------------------------------------------------------------
#define BB 8
#define LL 64
#define PP 128
#define EE 256   // VO=VR=TE=EOR=ERR

typedef __attribute__((ext_vector_type(16))) _Float16 v16h;
typedef __attribute__((ext_vector_type(4)))  _Float16 v4h;
typedef __attribute__((ext_vector_type(8)))  float    v8f;
typedef __attribute__((ext_vector_type(4)))  float    v4f;

__device__ __forceinline__ float silu_f(float x) {
    return x / (1.0f + __expf(-x));
}

// ---------------------------------------------------------------------------
// Weight prep: convert W[K,256] f32 row-major into f16 WMMA-B fragment layout.
// Fragment fb = (k>>5)*16 + (n>>4); within: lane = (n&15)|(((k>>3)&1)<<4),
// half = (k&7)|((k>>4)<<3).  Each thread handles 4 consecutive k for one n
// (same lane, 4 consecutive halves) -> one packed 8-byte store.
// grid.x = K/4, block = 256 (one thread per n).
// ---------------------------------------------------------------------------
__global__ __launch_bounds__(256)
void wprep_kernel(const float* __restrict__ W, _Float16* __restrict__ Wf)
{
    const int n  = threadIdx.x;
    const int k0 = blockIdx.x * 4;
    v4h h;
#pragma unroll
    for (int j = 0; j < 4; ++j)
        h[j] = (_Float16)W[(size_t)(k0 + j) * EE + n];
    const int fb = (k0 >> 5) * 16 + (n >> 4);
    const int ln = (n & 15) | (((k0 >> 3) & 1) << 4);
    const int hf = (k0 & 7) | (((k0 >> 4) & 1) << 3);
    *(v4h*)&Wf[((size_t)fb * 32 + ln) * 16 + hf] = h;
}

// ---------------------------------------------------------------------------
// WMMA GEMM: C[M,256] = epilogue( A[M,K] @ W[K,256] ), M multiple of 128.
//   x = acc (+bias[col]) (+add0[idx0*256+col]) (+add1[idx1*256+col]);
//   if act: x = silu(x); if post: x += post[row*256+col]
//   idx(row) = (((row>>sa)&ka)<<sm) + ((row>>sb)&kb)   (all pow2 broadcasts)
// Block tile 128x64, 8 waves, each wave 32x32 = 2x2 v_wmma_f32_16x16x32_f16.
// A staged via LDS (vector loads + packed b64 stores, fragment-permuted);
// B fragments read directly from pre-converted global f16 (hot in L2).
// ---------------------------------------------------------------------------
#define TM 128
#define TN 64
#define TK 32

__global__ __launch_bounds__(256, 2)
void gemm_wmma(const float* __restrict__ A, const _Float16* __restrict__ Wf,
               float* __restrict__ C, int M, int K,
               const float* __restrict__ bias,
               const float* __restrict__ add0, int s0a, int k0a, int s0m, int s0b, int k0b,
               const float* __restrict__ add1, int s1a, int k1a, int s1m, int s1b, int k1b,
               const float* __restrict__ post, int act)
{
    const int N    = EE;
    const int tid  = threadIdx.x;
    const int lane = tid & 31;
    const int w    = tid >> 5;
    const int wm   = w & 3;     // 4 row groups of 32
    const int wn   = w >> 2;    // 2 col groups of 32
    const int m0   = blockIdx.x * TM;
    const int n0   = blockIdx.y * TN;

    __shared__ v16h Asv[8][32];   // 8 subtiles of 16x32, fragment-major
    _Float16* Ash = (_Float16*)Asv;

    v8f acc[2][2] = {};

    // B fragment base for this wave (fragments of 512 halves; 16 per k-step)
    const _Float16* wfw = Wf + ((size_t)((n0 >> 4) + wn * 2) * 512) + (size_t)lane * 16;

    for (int kk = 0; kk < K; kk += TK) {
        __syncthreads();
        // ---- stage A tile (128x32): 1 float4 load + 1 packed v4h LDS store ----
#pragma unroll
        for (int i = 0; i < 4; ++i) {
            int e  = i * 256 + tid;
            int r  = e >> 3;            // row in tile
            int k0 = (e & 7) << 2;      // k group of 4
            v4f f = *(const v4f*)&A[(size_t)(m0 + r) * K + kk + k0];
            v4h h;
#pragma unroll
            for (int j = 0; j < 4; ++j) h[j] = (_Float16)f[j];
            int s  = r >> 4;
            int ln = (r & 15) | (((k0 >> 3) & 1) << 4);
            int hf = (k0 & 7) | ((k0 >> 4) << 3);
            *(v4h*)&Ash[(s * 32 + ln) * 16 + hf] = h;
        }
        __syncthreads();

        const _Float16* wp = wfw + (size_t)(kk >> 5) * (16 * 512);
        v16h b0 = *(const v16h*)(wp);
        v16h b1 = *(const v16h*)(wp + 512);
        v16h a0 = Asv[wm * 2 + 0][lane];
        v16h a1 = Asv[wm * 2 + 1][lane];
        acc[0][0] = __builtin_amdgcn_wmma_f32_16x16x32_f16(false, a0, false, b0, (short)0, acc[0][0], false, false);
        acc[0][1] = __builtin_amdgcn_wmma_f32_16x16x32_f16(false, a0, false, b1, (short)0, acc[0][1], false, false);
        acc[1][0] = __builtin_amdgcn_wmma_f32_16x16x32_f16(false, a1, false, b0, (short)0, acc[1][0], false, false);
        acc[1][1] = __builtin_amdgcn_wmma_f32_16x16x32_f16(false, a1, false, b1, (short)0, acc[1][1], false, false);
    }

    // ---- epilogue (C/D layout: col = lane&15, row = r + 8*(lane>>4)) ----
#pragma unroll
    for (int mi = 0; mi < 2; ++mi)
#pragma unroll
    for (int ni = 0; ni < 2; ++ni) {
        const float* af = (const float*)&acc[mi][ni];
        int col     = n0 + wn * 32 + ni * 16 + (lane & 15);
        int rowBase = m0 + wm * 32 + mi * 16 + ((lane >> 4) << 3);
#pragma unroll
        for (int r = 0; r < 8; ++r) {
            int row = rowBase + r;
            float x = af[r];
            if (bias) x += bias[col];
            if (add0) {
                int idx = (((row >> s0a) & k0a) << s0m) + ((row >> s0b) & k0b);
                x += add0[(size_t)idx * N + col];
            }
            if (add1) {
                int idx = (((row >> s1a) & k1a) << s1m) + ((row >> s1b) & k1b);
                x += add1[(size_t)idx * N + col];
            }
            if (act) x = silu_f(x);
            if (post) x += post[(size_t)row * N + col];
            C[(size_t)row * N + col] = x;
        }
    }
}

// ---------------------------------------------------------------------------
// Small matvec for M<=8 projections (t_embed / pool): out[row,n] = A[row]@W[:,n]
// grid.x = M, block = 256.
// ---------------------------------------------------------------------------
__global__ __launch_bounds__(256)
void matvec_kernel(const float* __restrict__ A, const float* __restrict__ W,
                   float* __restrict__ C, int K)
{
    const int row = blockIdx.x, n = threadIdx.x;
    float s = 0.0f;
    for (int k = 0; k < K; ++k)
        s += A[(size_t)row * K + k] * W[(size_t)k * EE + n];
    C[(size_t)row * EE + n] = s;
}

// ---------------------------------------------------------------------------
// Grouped attention + aggregation (H=4): per (b,l), logits over D keys x 64
// groups, softmax over keys, weighted sum of values.
// ---------------------------------------------------------------------------
template<int D>
__global__ __launch_bounds__(256)
void attn_agg_kernel(const float* __restrict__ q, const float* __restrict__ k,
                     const float* __restrict__ v, float* __restrict__ agg, int Lq)
{
    const int E = EE;
    __shared__ float sm[D * 64];
    const int bl = blockIdx.x;
    const int b  = bl / Lq;
    const int t  = threadIdx.x;
    const float scale = 0.28867513459481287f;  // 1/sqrt(H*3) = 1/sqrt(12)

    float qv = q[(size_t)bl * E + t];
    for (int p = 0; p < D; ++p) {
        float s = qv * k[((size_t)b * D + p) * E + t];
        s += __shfl_xor(s, 1);
        s += __shfl_xor(s, 2);
        if ((t & 3) == 0) sm[p * 64 + (t >> 2)] = s * scale;
    }
    __syncthreads();
    if (t < 64) {
        float mx = -3.4e38f;
        for (int p = 0; p < D; ++p) mx = fmaxf(mx, sm[p * 64 + t]);
        float sum = 0.0f;
        for (int p = 0; p < D; ++p) {
            float e = __expf(sm[p * 64 + t] - mx);
            sm[p * 64 + t] = e; sum += e;
        }
        float inv = 1.0f / sum;
        for (int p = 0; p < D; ++p) sm[p * 64 + t] *= inv;
    }
    __syncthreads();
    float a = 0.0f;
    const int g = t >> 2;
    for (int p = 0; p < D; ++p)
        a += sm[p * 64 + g] * v[((size_t)bl * D + p) * E + t];
    agg[(size_t)bl * E + t] = a;
}

// ---------------------------------------------------------------------------
// Mean / max reductions
// ---------------------------------------------------------------------------
__global__ __launch_bounds__(256)
void reduce_mean_kernel(const float* __restrict__ in, float* __restrict__ out,
                        int R, int innerN, int innerStride, int outerStride,
                        int strideR, float scale)
{
    const int o = blockIdx.x, e = threadIdx.x;
    size_t base = (size_t)(o / innerN) * outerStride + (size_t)(o % innerN) * innerStride;
    float s = 0.0f;
    for (int r = 0; r < R; ++r) s += in[(base + (size_t)r * strideR) * EE + e];
    out[(size_t)o * EE + e] = s * scale;
}

__global__ __launch_bounds__(256)
void maxpool_kernel(const float* __restrict__ in, float* __restrict__ out, int L_)
{
    const int b = blockIdx.x, e = threadIdx.x;
    float m = -3.4e38f;
    for (int l = 0; l < L_; ++l)
        m = fmaxf(m, in[((size_t)b * L_ + l) * EE + e]);
    out[(size_t)b * EE + e] = m;
}

// ---------------------------------------------------------------------------
// Host orchestration
// ---------------------------------------------------------------------------
static void gemm(hipStream_t st, const float* A, const float* W, _Float16* WF,
                 float* C, int M, int K, const float* bias,
                 const float* add0, int s0a, int k0a, int s0m, int s0b, int k0b,
                 const float* add1, int s1a, int k1a, int s1m, int s1b, int k1b,
                 const float* post, int act)
{
    wprep_kernel<<<dim3(K / 4), 256, 0, st>>>(W, WF);
    dim3 g(M / TM, EE / TN), blk(256);
    gemm_wmma<<<g, blk, 0, st>>>(A, WF, C, M, K, bias,
                                 add0, s0a, k0a, s0m, s0b, k0b,
                                 add1, s1a, k1a, s1m, s1b, k1b,
                                 post, act);
}

extern "C" void kernel_launch(void* const* d_in, const int* in_sizes, int n_in,
                              void* d_out, int out_size, void* d_ws, size_t ws_size,
                              hipStream_t stream)
{
    (void)in_sizes; (void)n_in; (void)out_size; (void)ws_size;
    // ---- inputs ----
    const float* object_node_f = (const float*)d_in[0];   // [B,P,256]
    const float* robot_node_f  = (const float*)d_in[1];   // [B,L,256]
    const float* or_edge_f     = (const float*)d_in[2];   // [B,L,P,256]
    const float* rr_edge_f     = (const float*)d_in[3];   // [B,L,L,256]
    const float* t_embed       = (const float*)d_in[4];   // [B,256]
    const float *vob_w=(const float*)d_in[5],  *vob_b=(const float*)d_in[6];
    const float *vrb_w=(const float*)d_in[7],  *vrb_b=(const float*)d_in[8];
    const float *eor_w=(const float*)d_in[9],  *eor_b=(const float*)d_in[10];
    const float *err_w=(const float*)d_in[11], *err_b=(const float*)d_in[12];
    const float *orq_w=(const float*)d_in[13], *orq_b=(const float*)d_in[14];
    const float *ork_w=(const float*)d_in[15], *ork_b=(const float*)d_in[16];
    const float *orv_w1=(const float*)d_in[17],*orv_b1=(const float*)d_in[18];
    const float *orv_w2=(const float*)d_in[19],*orv_b2=(const float*)d_in[20];
    const float *orself_w=(const float*)d_in[21],*orself_b=(const float*)d_in[22];
    const float *orout_w=(const float*)d_in[23], *orout_b=(const float*)d_in[24];
    const float *oredge_w1=(const float*)d_in[25],*oredge_b1=(const float*)d_in[26];
    const float *oredge_w2=(const float*)d_in[27],*oredge_b2=(const float*)d_in[28];
    const float *rrq_w=(const float*)d_in[29], *rrq_b=(const float*)d_in[30];
    const float *rrk_w=(const float*)d_in[31], *rrk_b=(const float*)d_in[32];
    const float *rrv_w1=(const float*)d_in[33],*rrv_b1=(const float*)d_in[34];
    const float *rrv_w2=(const float*)d_in[35],*rrv_b2=(const float*)d_in[36];
    const float *rrself_w=(const float*)d_in[37],*rrself_b=(const float*)d_in[38];
    const float *rrout_w=(const float*)d_in[39], *rrout_b=(const float*)d_in[40];
    const float *rredge_w1=(const float*)d_in[41],*rredge_b1=(const float*)d_in[42];
    const float *rredge_w2=(const float*)d_in[43],*rredge_b2=(const float*)d_in[44];

    const int SEG = EE * EE;   // 256-row slice stride in stacked weights

    // ---- workspace layout (floats) ----
    float* ws = (float*)d_ws;
    size_t off = 0;
    auto alloc = [&](size_t n){ float* p = ws + off; off += n; return p; };
    float* OBJ   = alloc((size_t)BB*PP*EE);
    float* ROB0  = alloc((size_t)BB*LL*EE);      // rob, then rob2
    float* ROB1  = alloc((size_t)BB*LL*EE);
    float* QB    = alloc((size_t)BB*LL*EE);
    float* KB    = alloc((size_t)BB*PP*EE);
    float* AGG   = alloc((size_t)BB*LL*EE);
    float* TEW   = alloc((size_t)BB*EE);
    float* S1    = alloc((size_t)BB*PP*EE);
    float* S2    = alloc((size_t)BB*LL*EE);
    float* S3    = alloc((size_t)BB*EE);
    float* POOL  = alloc((size_t)BB*EE);
    float* MEAN1 = alloc((size_t)BB*PP*EE);
    float* MEAN2 = alloc((size_t)BB*LL*EE);
    _Float16* WF = (_Float16*)alloc((size_t)EE*EE/2);  // f16 fragment weights
    float* BIGA  = alloc((size_t)BB*LL*PP*EE);   // ORE, later RRE
    float* BIGB  = alloc((size_t)BB*LL*PP*EE);   // ORV, later RRV
    float* HID   = alloc((size_t)BB*LL*PP*EE);   // shared MLP hidden

    // ---- outputs ----
    float* out_rob = (float*)d_out;                      // [B,L,256]
    float* out_or  = out_rob + (size_t)BB*LL*EE;         // [B,L,P,256]
    float* out_rr  = out_or  + (size_t)BB*LL*PP*EE;      // [B,L,L,256]

    const float* NUL = nullptr;
    const int Mor = BB*LL*PP, Mrr = BB*LL*LL;

    // ===================== bindings (or phase) =====================
    matvec_kernel<<<BB, 256, 0, stream>>>(t_embed, vob_w + SEG, TEW, EE);
    gemm(stream, object_node_f, vob_w, WF, OBJ, BB*PP, EE, vob_b,
         TEW, 7,7,0, 0,0, NUL,0,0,0,0,0, NUL, 1);                       // idx=b
    matvec_kernel<<<BB, 256, 0, stream>>>(t_embed, vrb_w + SEG, TEW, EE);
    gemm(stream, robot_node_f, vrb_w, WF, ROB0, BB*LL, EE, vrb_b,
         TEW, 6,7,0, 0,0, NUL,0,0,0,0,0, NUL, 1);
    matvec_kernel<<<BB, 256, 0, stream>>>(t_embed, eor_w + SEG, TEW, EE);
    gemm(stream, or_edge_f, eor_w, WF, BIGA, Mor, EE, eor_b,            // ORE
         TEW, 13,7,0, 0,0, NUL,0,0,0,0,0, NUL, 1);

    // ===================== object -> robot attention =====================
    gemm(stream, ROB0, orq_w, WF, QB, BB*LL, EE, orq_b,
         NUL,0,0,0,0,0, NUL,0,0,0,0,0, NUL, 0);
    gemm(stream, OBJ, ork_w, WF, KB, BB*PP, EE, ork_b,
         NUL,0,0,0,0,0, NUL,0,0,0,0,0, NUL, 0);
    // or_value = mlp2(concat[obj, rob, ore]) via segment GEMMs
    gemm(stream, OBJ,  orv_w1,       WF, S1, BB*PP, EE, NUL,
         NUL,0,0,0,0,0, NUL,0,0,0,0,0, NUL, 0);
    gemm(stream, ROB0, orv_w1 + SEG, WF, S2, BB*LL, EE, NUL,
         NUL,0,0,0,0,0, NUL,0,0,0,0,0, NUL, 0);
    gemm(stream, BIGA, orv_w1 + 2*SEG, WF, HID, Mor, EE, orv_b1,
         S1, 13,7,7, 0,127,            // idx = b*128 + p
         S2, 13,7,6, 7,63,             // idx = b*64  + l
         NUL, 1);
    gemm(stream, HID, orv_w2, WF, BIGB, Mor, EE, orv_b2,                // ORV
         NUL,0,0,0,0,0, NUL,0,0,0,0,0, NUL, 0);
    attn_agg_kernel<PP><<<BB*LL, 256, 0, stream>>>(QB, KB, BIGB, AGG, LL);
    gemm(stream, ROB0, orself_w, WF, ROB1, BB*LL, EE, orself_b,
         NUL,0,0,0,0,0, NUL,0,0,0,0,0, AGG, 1);
    maxpool_kernel<<<BB, 256, 0, stream>>>(ROB1, POOL, LL);
    matvec_kernel<<<BB, 256, 0, stream>>>(POOL, orout_w + SEG, S3, EE);
    gemm(stream, ROB1, orout_w, WF, ROB0, BB*LL, EE, orout_b,           // rob2
         S3, 6,7,0, 0,0, NUL,0,0,0,0,0, NUL, 1);
    // or_pool means (0.5 baked in) projected through oredge_w1[256:512]
    reduce_mean_kernel<<<BB*PP, 256, 0, stream>>>(BIGA, MEAN1, LL, PP, 1, LL*PP, PP, 0.5f/LL);
    reduce_mean_kernel<<<BB*LL, 256, 0, stream>>>(BIGA, MEAN2, PP, LL, PP, LL*PP, 1, 0.5f/PP);
    gemm(stream, MEAN1, oredge_w1 + SEG, WF, S1, BB*PP, EE, NUL,
         NUL,0,0,0,0,0, NUL,0,0,0,0,0, NUL, 0);
    gemm(stream, MEAN2, oredge_w1 + SEG, WF, S2, BB*LL, EE, NUL,
         NUL,0,0,0,0,0, NUL,0,0,0,0,0, NUL, 0);
    gemm(stream, BIGB, oredge_w1, WF, HID, Mor, EE, oredge_b1,
         S1, 13,7,7, 0,127,
         S2, 13,7,6, 7,63,
         NUL, 1);
    gemm(stream, HID, oredge_w2, WF, out_or, Mor, EE, oredge_b2,
         NUL,0,0,0,0,0, NUL,0,0,0,0,0, NUL, 0);

    // ===================== rr binding (deferred; reuses BIGA) =====================
    matvec_kernel<<<BB, 256, 0, stream>>>(t_embed, err_w + SEG, TEW, EE);
    gemm(stream, rr_edge_f, err_w, WF, BIGA, Mrr, EE, err_b,            // RRE
         TEW, 12,7,0, 0,0, NUL,0,0,0,0,0, NUL, 1);

    // ===================== robot -> robot attention =====================
    gemm(stream, ROB0, rrq_w, WF, QB, BB*LL, EE, rrq_b,
         NUL,0,0,0,0,0, NUL,0,0,0,0,0, NUL, 0);
    gemm(stream, ROB0, rrk_w, WF, KB, BB*LL, EE, rrk_b,
         NUL,0,0,0,0,0, NUL,0,0,0,0,0, NUL, 0);
    gemm(stream, ROB0, rrv_w1,       WF, S1, BB*LL, EE, NUL,
         NUL,0,0,0,0,0, NUL,0,0,0,0,0, NUL, 0);
    gemm(stream, ROB0, rrv_w1 + SEG, WF, S2, BB*LL, EE, NUL,
         NUL,0,0,0,0,0, NUL,0,0,0,0,0, NUL, 0);
    gemm(stream, BIGA, rrv_w1 + 2*SEG, WF, HID, Mrr, EE, rrv_b1,
         S1, 12,7,6, 0,63,             // es: idx = b*64 + j
         S2, 12,7,6, 6,63,             // en: idx = b*64 + i
         NUL, 1);
    gemm(stream, HID, rrv_w2, WF, BIGB, Mrr, EE, rrv_b2,                // RRV
         NUL,0,0,0,0,0, NUL,0,0,0,0,0, NUL, 0);
    attn_agg_kernel<LL><<<BB*LL, 256, 0, stream>>>(QB, KB, BIGB, AGG, LL);
    gemm(stream, ROB0, rrself_w, WF, ROB1, BB*LL, EE, rrself_b,
         NUL,0,0,0,0,0, NUL,0,0,0,0,0, AGG, 1);
    maxpool_kernel<<<BB, 256, 0, stream>>>(ROB1, POOL, LL);
    matvec_kernel<<<BB, 256, 0, stream>>>(POOL, rrout_w + SEG, S3, EE);
    gemm(stream, ROB1, rrout_w, WF, out_rob, BB*LL, EE, rrout_b,        // final rob
         S3, 6,7,0, 0,0, NUL,0,0,0,0,0, NUL, 1);
    // rr_pool means projected through rredge_w1[256:512]
    reduce_mean_kernel<<<BB*LL, 256, 0, stream>>>(BIGA, MEAN1, LL, LL, 1, LL*LL, LL, 0.5f/LL);
    reduce_mean_kernel<<<BB*LL, 256, 0, stream>>>(BIGA, MEAN2, LL, LL, LL, LL*LL, 1, 0.5f/LL);
    gemm(stream, MEAN1, rredge_w1 + SEG, WF, S1, BB*LL, EE, NUL,
         NUL,0,0,0,0,0, NUL,0,0,0,0,0, NUL, 0);
    gemm(stream, MEAN2, rredge_w1 + SEG, WF, S2, BB*LL, EE, NUL,
         NUL,0,0,0,0,0, NUL,0,0,0,0,0, NUL, 0);
    gemm(stream, BIGB, rredge_w1, WF, HID, Mrr, EE, rredge_b1,
         S1, 12,7,6, 0,63,
         S2, 12,7,6, 6,63,
         NUL, 1);
    gemm(stream, HID, rredge_w2, WF, out_rr, Mrr, EE, rredge_b2,
         NUL,0,0,0,0,0, NUL,0,0,0,0,0, NUL, 0);
}
